// OutputModule_68650757259667
// MI455X (gfx1250) — compile-verified, matrix operand order checked by reading, not run on hardware
//
#include <hip/hip_runtime.h>
#include <hip/hip_bf16.h>
#include <math.h>

// ---------------- problem constants (from reference) ----------------
#define BG    8            // graphs
#define LL    8000         // tokens per graph
#define DD    512          // embed dim
#define NPG   1000         // nodes per graph
#define EPG   7000         // edge tokens per graph
#define NNODE 8000         // total nodes
#define NEDGE 56000        // total edges
#define FFD   1024         // hidden dim
#define NLH   3            // residual hidden layers
#define NGAU  50           // gaussians
#define K1    1536         // 3*D first-layer K

typedef __attribute__((ext_vector_type(8)))  _Float16 v8h;
typedef __attribute__((ext_vector_type(16))) _Float16 v16h;
typedef __attribute__((ext_vector_type(8)))  float    v8f;

__device__ __forceinline__ v16h concat8(v8h lo, v8h hi) {
  return __builtin_shufflevector(lo, hi, 0,1,2,3,4,5,6,7,8,9,10,11,12,13,14,15);
}

__device__ __forceinline__ float gelu_tanh(float v) {
  const float c0 = 0.7978845608028654f;   // sqrt(2/pi)
  float t = c0 * (v + 0.044715f * v * v * v);
  return 0.5f * v * (1.0f + tanhf(t));
}

// ---------------- utility kernels ----------------
__global__ void zero_f32(float* p, int n) {
  for (int i = blockIdx.x * blockDim.x + threadIdx.x; i < n; i += gridDim.x * blockDim.x)
    p[i] = 0.0f;
}

__global__ void cvt_f32_f16(const float* __restrict__ src, _Float16* __restrict__ dst, int n) {
  for (int i = blockIdx.x * blockDim.x + threadIdx.x; i < n; i += gridDim.x * blockDim.x)
    dst[i] = (_Float16)src[i];
}

// ---------------- fused WMMA GEMM ----------------
// Out[M,N] = mode(A[M,K] * W[N,K]^T + bias[N])
//   mode 0: out = acc + b
//   mode 1: out = gelu(acc + b)
//   mode 2: out = Res + gelu(acc + b)     (Res == layer input, residual block)
// If A == nullptr: layer-1 gather mode — A rows are built from x/edge_index:
//   k in [0,512):    nodes[src[e]]   row in x:  (n%NPG)*B + n/NPG
//   k in [512,1024): nodes[dst[e]]
//   k in [1024,1536): edges[e]       row in x:  (NPG + e%EPG)*B + e/EPG
#define LDS_STRIDE 72   // 64 K-halves + pad (keeps 16B alignment, breaks bank stride)

__global__ __launch_bounds__(256)
void gemm_wmma_f16(const _Float16* __restrict__ A,
                   const _Float16* __restrict__ W,
                   const float*    __restrict__ bias,
                   const _Float16* __restrict__ Res,
                   _Float16*       __restrict__ Out,
                   int M, int N, int K, int mode,
                   const float* __restrict__ x,
                   const int*   __restrict__ ei) {
  __shared__ _Float16 smA[128 * LDS_STRIDE];

  const int tid   = threadIdx.x;
  const int lane  = tid & 31;
  const int wave  = tid >> 5;          // 0..7
  const int wM    = wave >> 2;         // 0..1  (64 rows each)
  const int wN    = wave & 3;          // 0..3  (32 cols each)
  const int lhi   = lane >> 4;         // K-half selector for 16-bit operands
  const int l15   = lane & 15;

  const int blockRow0 = blockIdx.x * 128;
  const int blockCol0 = blockIdx.y * 128;

  // staging assignment: 2 threads per row, 32 halves each
  const int srow  = tid >> 1;          // 0..127
  const int shalf = tid & 1;           // which 32-half chunk
  const int gr    = blockRow0 + srow;  // global row for staging

  // per-lane weight-column base pointers (B operand streams straight from L2)
  const _Float16* wp[2];
  int col[2];
#pragma unroll
  for (int nt = 0; nt < 2; ++nt) {
    col[nt] = blockCol0 + wN * 32 + nt * 16 + l15;
    wp[nt]  = W + (size_t)col[nt] * K;
  }

  v8f acc[8];
#pragma unroll
  for (int i = 0; i < 8; ++i) acc[i] = (v8f){0.f,0.f,0.f,0.f,0.f,0.f,0.f,0.f};

  for (int kb = 0; kb < K; kb += 64) {
    __syncthreads();
    // ---- stage A[128 x 64] into LDS (f16) ----
    _Float16* dst = &smA[srow * LDS_STRIDE + shalf * 32];
    if (A != nullptr) {
      if (gr < M) {
        const v8h* s = (const v8h*)(A + (size_t)gr * K + kb + shalf * 32);
        v8h* dv = (v8h*)dst;
        dv[0] = s[0]; dv[1] = s[1]; dv[2] = s[2]; dv[3] = s[3];
      } else {
        v8h z = (v8h){0,0,0,0,0,0,0,0};
        v8h* dv = (v8h*)dst;
        dv[0] = z; dv[1] = z; dv[2] = z; dv[3] = z;
      }
    } else {
      // gather + f32->f16 convert (layer 1)
      if (gr < M) {
        const int seg = kb >> 9;                  // 0:src 1:dst 2:edge
        const int d0  = (kb & 511) + shalf * 32;  // within 512-wide segment
        int xrow;
        if (seg == 0)      { int n = ei[gr];         xrow = (n % NPG) * BG + n / NPG; }
        else if (seg == 1) { int n = ei[NEDGE + gr]; xrow = (n % NPG) * BG + n / NPG; }
        else               { xrow = (NPG + gr % EPG) * BG + gr / EPG; }
        const float* src = x + (size_t)xrow * DD + d0;
        float tmp[32];
#pragma unroll
        for (int j = 0; j < 32; j += 4) {
          const float4 v = *(const float4*)(src + j);
          tmp[j] = v.x; tmp[j+1] = v.y; tmp[j+2] = v.z; tmp[j+3] = v.w;
        }
#pragma unroll
        for (int j = 0; j < 32; ++j) dst[j] = (_Float16)tmp[j];
      } else {
        v8h z = (v8h){0,0,0,0,0,0,0,0};
        v8h* dv = (v8h*)dst;
        dv[0] = z; dv[1] = z; dv[2] = z; dv[3] = z;
      }
    }
    __syncthreads();

    // ---- compute: 2 K-chunks of 32, 8 WMMAs each ----
#pragma unroll
    for (int kc = 0; kc < 64; kc += 32) {
      v16h afr[4];
#pragma unroll
      for (int mt = 0; mt < 4; ++mt) {
        const _Float16* ap = &smA[(wM * 64 + mt * 16 + l15) * LDS_STRIDE + kc + lhi * 8];
        v8h lo = *(const v8h*)ap;
        v8h hi = *(const v8h*)(ap + 16);
        afr[mt] = concat8(lo, hi);
      }
#pragma unroll
      for (int nt = 0; nt < 2; ++nt) {
        const _Float16* bp = wp[nt] + kb + kc + lhi * 8;
        v8h lo = *(const v8h*)bp;
        v8h hi = *(const v8h*)(bp + 16);
        v16h bfr = concat8(lo, hi);
#pragma unroll
        for (int mt = 0; mt < 4; ++mt)
          acc[mt * 2 + nt] = __builtin_amdgcn_wmma_f32_16x16x32_f16(
              false, afr[mt], false, bfr, (short)0, acc[mt * 2 + nt], false, false);
      }
    }
  }

  // ---- epilogue: bias + activation (+residual), f32 -> f16 store ----
#pragma unroll
  for (int mt = 0; mt < 4; ++mt) {
#pragma unroll
    for (int nt = 0; nt < 2; ++nt) {
      const int c  = col[nt];
      const float bb = bias[c];
      const int row0 = blockRow0 + wM * 64 + mt * 16 + lhi * 8;
      v8f a = acc[mt * 2 + nt];
#pragma unroll
      for (int r = 0; r < 8; ++r) {
        const int row = row0 + r;
        if (row < M) {
          float v = a[r] + bb;
          if (mode >= 1) v = gelu_tanh(v);
          if (mode == 2) v += (float)Res[(size_t)row * N + c];
          Out[(size_t)row * N + c] = (_Float16)v;
        }
      }
    }
  }
}

// ---------------- per-edge epilogue: RBF weighting + scatter reductions ----------------
// one wave (32 lanes) per edge; OutM is [NE, 512] f16
__global__ __launch_bounds__(256)
void edge_epilogue(const _Float16* __restrict__ OutM,
                   const float* __restrict__ rbf_w,   // [512, 50]
                   const float* __restrict__ out_w,   // [512]
                   const float* __restrict__ pos,     // [NN, 3]
                   const int*   __restrict__ ei,      // [2, NE]
                   float* __restrict__ dOut,          // [8 energy | 24000 forces]
                   int isForce) {
  const int lane = threadIdx.x & 31;
  const int e    = blockIdx.x * (blockDim.x >> 5) + (threadIdx.x >> 5);
  if (e >= NEDGE) return;

  const int s = ei[e];
  const int d = ei[NEDGE + e];
  const float dx = pos[s * 3 + 0] - pos[d * 3 + 0];
  const float dy = pos[s * 3 + 1] - pos[d * 3 + 1];
  const float dz = pos[s * 3 + 2] - pos[d * 3 + 2];
  const float dist = sqrtf(dx * dx + dy * dy + dz * dz);

  // GaussianSmearing: 50 centers on [0, 12]
  const float delta = 12.0f / 49.0f;
  const float coeff = -0.5f / (delta * delta);
  float gv[NGAU];
#pragma unroll
  for (int g = 0; g < NGAU; ++g) {
    const float t = dist - (float)g * delta;
    gv[g] = __expf(coeff * t * t);
  }

  float partial = 0.0f;
  for (int j = 0; j < 16; ++j) {
    const int dd = lane + j * 32;
    const float w = out_w[dd];
    const float* rw = rbf_w + dd * NGAU;
    float ssum = 0.0f;
#pragma unroll
    for (int g = 0; g < NGAU; ++g) ssum += gv[g] * rw[g];
    partial += (float)OutM[(size_t)e * 512 + dd] * w * ssum;
  }
#pragma unroll
  for (int off = 16; off > 0; off >>= 1)
    partial += __shfl_xor(partial, off, 32);

  if (lane == 0) {
    if (!isForce) {
      atomicAdd(dOut + (e / EPG), partial);           // energy per graph
    } else {
      const float inv = 1.0f / fmaxf(dist, 1e-12f);
      atomicAdd(dOut + 8 + s * 3 + 0, partial * dx * inv);
      atomicAdd(dOut + 8 + s * 3 + 1, partial * dy * inv);
      atomicAdd(dOut + 8 + s * 3 + 2, partial * dz * inv);
    }
  }
}

// ---------------- host side ----------------
extern "C" void kernel_launch(void* const* d_in, const int* in_sizes, int n_in,
                              void* d_out, int out_size, void* d_ws, size_t ws_size,
                              hipStream_t stream) {
  (void)in_sizes; (void)n_in; (void)ws_size;

  const float* x       = (const float*)d_in[0];
  const float* pos     = (const float*)d_in[1];
  const int*   ei      = (const int*)d_in[3];
  const float* rbf_w_m[2] = { (const float*)d_in[6], (const float*)d_in[8] };
  const float* out_w_m[2] = { (const float*)d_in[7], (const float*)d_in[9] };
  const float* w_in_m [2] = { (const float*)d_in[10], (const float*)d_in[16] };
  const float* b_in_m [2] = { (const float*)d_in[11], (const float*)d_in[17] };
  const float* w_hid_m[2] = { (const float*)d_in[12], (const float*)d_in[18] };
  const float* b_hid_m[2] = { (const float*)d_in[13], (const float*)d_in[19] };
  const float* w_out_m[2] = { (const float*)d_in[14], (const float*)d_in[20] };
  const float* b_out_m[2] = { (const float*)d_in[15], (const float*)d_in[21] };
  float* out = (float*)d_out;

  // workspace layout (all offsets 256B-aligned)
  char* ws = (char*)d_ws;
  size_t off = 0;
  _Float16* H0 = (_Float16*)(ws + off); off += (size_t)NEDGE * FFD * 2;   // 114,688,000
  _Float16* H1 = (_Float16*)(ws + off); off += (size_t)NEDGE * FFD * 2;
  _Float16* W16_in [2]; _Float16* W16_hid[2]; _Float16* W16_out[2];
  for (int m = 0; m < 2; ++m) {
    W16_in [m] = (_Float16*)(ws + off); off += (size_t)FFD * K1  * 2;       // 3,145,728
    W16_hid[m] = (_Float16*)(ws + off); off += (size_t)NLH * FFD * FFD * 2; // 6,291,456
    W16_out[m] = (_Float16*)(ws + off); off += (size_t)DD  * FFD * 2;       // 1,048,576
  }

  // 1) zero outputs (harness poisons d_out)
  zero_f32<<<64, 256, 0, stream>>>(out, out_size);

  // 2) weights f32 -> f16
  for (int m = 0; m < 2; ++m) {
    cvt_f32_f16<<<2048, 256, 0, stream>>>(w_in_m[m],  W16_in [m], FFD * K1);
    cvt_f32_f16<<<2048, 256, 0, stream>>>(w_hid_m[m], W16_hid[m], NLH * FFD * FFD);
    cvt_f32_f16<<<2048, 256, 0, stream>>>(w_out_m[m], W16_out[m], DD * FFD);
  }

  const int gm = (NEDGE + 127) / 128;   // 438 row-tiles
  for (int m = 0; m < 2; ++m) {
    // layer 1: fused gather + GEMM + gelu  -> H0
    gemm_wmma_f16<<<dim3(gm, FFD / 128), 256, 0, stream>>>(
        nullptr, W16_in[m], b_in_m[m], nullptr, H0,
        NEDGE, FFD, K1, 1, x, ei);
    // 3 residual hidden blocks: h = h + gelu(h @ Wi^T + bi)
    _Float16* hin = H0; _Float16* hout = H1;
    for (int i = 0; i < NLH; ++i) {
      gemm_wmma_f16<<<dim3(gm, FFD / 128), 256, 0, stream>>>(
          hin, W16_hid[m] + (size_t)i * FFD * FFD, b_hid_m[m] + i * FFD, hin, hout,
          NEDGE, FFD, FFD, 2, nullptr, nullptr);
      _Float16* t = hin; hin = hout; hout = t;
    }
    // output projection (no activation): hin -> H-scratch [NE, 512]
    gemm_wmma_f16<<<dim3(gm, DD / 128), 256, 0, stream>>>(
        hin, W16_out[m], b_out_m[m], nullptr, hout,
        NEDGE, DD, FFD, 0, nullptr, nullptr);
    // RBF weighting + scatter (energy for m==0, forces for m==1)
    edge_epilogue<<<(NEDGE + 7) / 8, 256, 0, stream>>>(
        hout, rbf_w_m[m], out_w_m[m], pos, ei, out, m);
  }
}
